// MultiHeadLatentAttention_40827959116413
// MI455X (gfx1250) — compile-verified, matrix operand order checked by reading, not run on hardware
//
#include <hip/hip_runtime.h>

// ---------------------------------------------------------------------------
// MLA attention for MI455X (gfx1250, wave32, WMMA + Tensor Data Mover).
// B=2, S=2048, D=2048, H=16, KVH=4, HD=128, L=512.
// GEMMs + attention matmuls: v_wmma_f32_16x16x32_f16 (f32 accumulate).
// GEMM A/B tile staging: TDM tensor_load_to_lds, double buffered, with
// hardware LDS padding (64B interval + 8B pad == 72B row stride).
// ---------------------------------------------------------------------------

typedef _Float16 h16;
typedef __attribute__((ext_vector_type(16))) _Float16 v16h;
typedef __attribute__((ext_vector_type(8)))  float    v8f;
typedef __attribute__((ext_vector_type(4)))  unsigned int v4u;
typedef __attribute__((ext_vector_type(4)))  int v4i;
typedef __attribute__((ext_vector_type(8)))  int v8i;

#if defined(__has_builtin)
#if __has_builtin(__builtin_amdgcn_tensor_load_to_lds) && \
    __has_builtin(__builtin_amdgcn_s_wait_tensorcnt)
#define HAVE_TDM 1
#endif
#endif
#ifndef HAVE_TDM
#define HAVE_TDM 0
#endif

union Frag {
    v16h v;
    unsigned long long q[4];
};

__device__ inline v8f v8f_zero() {
    v8f z;
#pragma unroll
    for (int i = 0; i < 8; ++i) z[i] = 0.0f;
    return z;
}

static constexpr int LSTR = 36;  // LDS row stride in h16 (32 data + 4 pad = 72B)

#if HAVE_TDM
// Issue one TDM 2D tile load: 32 h16 x `rows` rows, row stride `k_elems` h16,
// into LDS at byte address `lds_addr` with 8B pad after every 64B row
// (-> 72B LDS row stride, matching LSTR).
__device__ inline void tdm_load_tile(const h16* gsrc, unsigned lds_addr,
                                     unsigned k_elems, unsigned rows) {
    unsigned long long ga = (unsigned long long)(uintptr_t)(const void*)gsrc;
    v4u g0;
    g0[0] = 1u;                                   // count=1 (valid), user mode
    g0[1] = lds_addr;                             // lds_addr [63:32]
    g0[2] = (unsigned)(ga & 0xffffffffu);         // global_addr [95:64]
    g0[3] = (unsigned)((ga >> 32) & 0x01ffffffu)  // global_addr [120:96]
          | (2u << 30);                           // type=2 ("image")
    v8i g1;
    // data_size=1 (2B), pad_enable, pad_interval=3 (64B), pad_amount=1 (8B)
    g1[0] = (int)((1u << 16) | (1u << 20) | (3u << 22) | (1u << 25));
    g1[1] = (int)((k_elems & 0xffffu) << 16);               // tensor_dim0 lo16
    g1[2] = (int)(((k_elems >> 16) & 0xffffu) |             // tensor_dim0 hi16
                  ((rows & 0xffffu) << 16));                // tensor_dim1 lo16
    g1[3] = (int)(((rows >> 16) & 0xffffu) | (32u << 16));  // tdim1 hi | tile_dim0=32
    g1[4] = (int)(rows & 0xffffu);                          // tile_dim1=rows, tile_dim2=0
    g1[5] = (int)k_elems;                                   // tensor_dim0_stride lo32
    g1[6] = 0;                                              // stride hi | dim1_stride lo
    g1[7] = 0;
    v4i gz;
    gz[0] = 0; gz[1] = 0; gz[2] = 0; gz[3] = 0;
#if __clang_major__ >= 23
    v8i gz8;
#pragma unroll
    for (int i = 0; i < 8; ++i) gz8[i] = 0;
    __builtin_amdgcn_tensor_load_to_lds(g0, g1, gz, gz, gz8, 0);
#else
    __builtin_amdgcn_tensor_load_to_lds(g0, g1, gz, gz, 0);
#endif
}
#endif

// ---------------------------------------------------------------------------
// Elementwise casts / transposes
// ---------------------------------------------------------------------------
__global__ void cast_f32_h16(const float* __restrict__ X, h16* __restrict__ Y, long n) {
    long i = (long)blockIdx.x * blockDim.x + threadIdx.x;
    if (i < n) Y[i] = (h16)X[i];
}

// W [K][N] f32 -> Wt [N][K] h16
__global__ void transpose_f32_h16(const float* __restrict__ W, h16* __restrict__ Wt,
                                  int K, int N) {
    long idx = (long)blockIdx.x * blockDim.x + threadIdx.x;
    if (idx >= (long)K * N) return;
    int n = (int)(idx % N);
    int k = (int)(idx / N);
    Wt[(long)n * K + k] = (h16)W[idx];
}

// ---------------------------------------------------------------------------
// Generic f16 WMMA GEMM:  C[M][N] = A[M][K] @ Bt[N][K]^T
// Block tile 128x128 (8 waves, each 32x64 = 2x4 16x16 tiles), K-step 32.
// TDM path: wave 0 streams double-buffered A/B tiles via tensor_load_to_lds.
// ---------------------------------------------------------------------------
template <bool F32OUT>
__global__ __launch_bounds__(256) void gemm_wmma_f16(
    const h16* __restrict__ A, const h16* __restrict__ Bt,
    void* __restrict__ Cout, int M, int N, int K) {
    const int tid  = threadIdx.x;
    const int lane = tid & 31;
    const int wave = tid >> 5;
    const int wm = (wave >> 1) * 32;
    const int wn = (wave & 1) * 64;
    const long bm = (long)blockIdx.x * 128;
    const long bn = (long)blockIdx.y * 128;
    const int r  = lane & 15;
    const int hi = lane >> 4;
    const int c0 = hi * 8;  // ISA 16-bit A-frag K interleave

    v8f acc[2][4];
#pragma unroll
    for (int mi = 0; mi < 2; ++mi)
#pragma unroll
        for (int ni = 0; ni < 4; ++ni) acc[mi][ni] = v8f_zero();

#if HAVE_TDM
    __shared__ __attribute__((aligned(128))) h16 As[2][128 * LSTR];
    __shared__ __attribute__((aligned(128))) h16 Bs[2][128 * LSTR];
    const h16* Abase = A + bm * K;
    const h16* Bbase = Bt + bn * K;
    const unsigned asb = (unsigned)(uintptr_t)(void*)&As[0][0];
    const unsigned bsb = (unsigned)(uintptr_t)(void*)&Bs[0][0];
    const unsigned bufsz = 128 * LSTR * 2;  // bytes per buffer

    const int nk = K >> 5;
    if (wave == 0) {  // prologue: tile 0 in flight
        tdm_load_tile(Abase, asb, (unsigned)K, 128u);
        tdm_load_tile(Bbase, bsb, (unsigned)K, 128u);
    }
    for (int t = 0; t < nk; ++t) {
        const int buf = t & 1;
        if (wave == 0) {
            if (t + 1 < nk) {  // prefetch next tile into other buffer
                tdm_load_tile(Abase + (long)(t + 1) * 32, asb + (buf ^ 1) * bufsz,
                              (unsigned)K, 128u);
                tdm_load_tile(Bbase + (long)(t + 1) * 32, bsb + (buf ^ 1) * bufsz,
                              (unsigned)K, 128u);
                __builtin_amdgcn_s_wait_tensorcnt(2);  // tile t landed
            } else {
                __builtin_amdgcn_s_wait_tensorcnt(0);
            }
        }
        __syncthreads();

        Frag af[2], bf[4];
#pragma unroll
        for (int mi = 0; mi < 2; ++mi) {
            const h16* rowp = &As[buf][(wm + mi * 16 + r) * LSTR];
            const unsigned long long* q0 = (const unsigned long long*)(rowp + c0);
            const unsigned long long* q1 = (const unsigned long long*)(rowp + c0 + 16);
            af[mi].q[0] = q0[0]; af[mi].q[1] = q0[1];
            af[mi].q[2] = q1[0]; af[mi].q[3] = q1[1];
        }
#pragma unroll
        for (int ni = 0; ni < 4; ++ni) {
            const h16* rowp = &Bs[buf][(wn + ni * 16 + r) * LSTR];
            const unsigned long long* q0 = (const unsigned long long*)(rowp + c0);
            const unsigned long long* q1 = (const unsigned long long*)(rowp + c0 + 16);
            bf[ni].q[0] = q0[0]; bf[ni].q[1] = q0[1];
            bf[ni].q[2] = q1[0]; bf[ni].q[3] = q1[1];
        }
#pragma unroll
        for (int mi = 0; mi < 2; ++mi)
#pragma unroll
            for (int ni = 0; ni < 4; ++ni)
                acc[mi][ni] = __builtin_amdgcn_wmma_f32_16x16x32_f16(
                    false, af[mi].v, false, bf[ni].v, (short)0, acc[mi][ni],
                    false, false);
        __syncthreads();  // all waves done with buf before TDM refills it
    }
#else
    // Fallback: per-thread synchronous staging (round-1 path)
    __shared__ __attribute__((aligned(16))) h16 As[128 * LSTR];
    __shared__ __attribute__((aligned(16))) h16 Bs[128 * LSTR];
    const int lrow = tid >> 1;
    const int lseg = (tid & 1) * 16;
    for (long k0 = 0; k0 < K; k0 += 32) {
        ulonglong2 av = *(const ulonglong2*)(A + (bm + lrow) * K + k0 + lseg);
        ulonglong2 bv = *(const ulonglong2*)(Bt + (bn + lrow) * K + k0 + lseg);
        unsigned long long* pa = (unsigned long long*)(As + lrow * LSTR + lseg);
        pa[0] = av.x; pa[1] = av.y;
        unsigned long long* pb = (unsigned long long*)(Bs + lrow * LSTR + lseg);
        pb[0] = bv.x; pb[1] = bv.y;
        __syncthreads();
        Frag af[2], bf[4];
#pragma unroll
        for (int mi = 0; mi < 2; ++mi) {
            const h16* rowp = As + (wm + mi * 16 + r) * LSTR;
            const unsigned long long* q0 = (const unsigned long long*)(rowp + c0);
            const unsigned long long* q1 = (const unsigned long long*)(rowp + c0 + 16);
            af[mi].q[0] = q0[0]; af[mi].q[1] = q0[1];
            af[mi].q[2] = q1[0]; af[mi].q[3] = q1[1];
        }
#pragma unroll
        for (int ni = 0; ni < 4; ++ni) {
            const h16* rowp = Bs + (wn + ni * 16 + r) * LSTR;
            const unsigned long long* q0 = (const unsigned long long*)(rowp + c0);
            const unsigned long long* q1 = (const unsigned long long*)(rowp + c0 + 16);
            bf[ni].q[0] = q0[0]; bf[ni].q[1] = q0[1];
            bf[ni].q[2] = q1[0]; bf[ni].q[3] = q1[1];
        }
#pragma unroll
        for (int mi = 0; mi < 2; ++mi)
#pragma unroll
            for (int ni = 0; ni < 4; ++ni)
                acc[mi][ni] = __builtin_amdgcn_wmma_f32_16x16x32_f16(
                    false, af[mi].v, false, bf[ni].v, (short)0, acc[mi][ni],
                    false, false);
        __syncthreads();
    }
#endif

    // ---- branch-free epilogue ----
#pragma unroll
    for (int mi = 0; mi < 2; ++mi)
#pragma unroll
        for (int ni = 0; ni < 4; ++ni)
#pragma unroll
            for (int j = 0; j < 8; ++j) {
                long row = bm + wm + mi * 16 + j + hi * 8;
                long col = bn + wn + ni * 16 + r;
                float v = acc[mi][ni][j];
                if constexpr (F32OUT) ((float*)Cout)[row * N + col] = v;
                else                  ((h16*)Cout)[row * N + col] = (h16)v;
            }
}

// ---------------------------------------------------------------------------
// RoPE (rotate-half) for Q; write (b,h,s,hd) h16
// ---------------------------------------------------------------------------
__global__ void rope_q_kernel(const float* __restrict__ q, const float* __restrict__ cs,
                              const float* __restrict__ sn, h16* __restrict__ Qh) {
    long idx = (long)blockIdx.x * blockDim.x + threadIdx.x;  // B*S*H*HD
    if (idx >= (long)2 * 2048 * 16 * 128) return;
    int i = (int)(idx & 127);
    int h = (int)((idx >> 7) & 15);
    int s = (int)((idx >> 11) & 2047);
    int b = (int)(idx >> 22);
    long src = ((long)(b * 2048 + s)) * 2048 + h * 128 + i;
    float c = cs[s * 128 + i], sv = sn[s * 128 + i];
    float v = q[src];
    float rot = (i < 64) ? -q[src + 64] : q[src - 64];
    Qh[(((long)b * 16 + h) * 2048 + s) * 128 + i] = (h16)(v * c + rot * sv);
}

// RoPE for K -> (b,kvh,s,hd); V -> transposed (b,kvh,hd,s)
__global__ void rope_kv_kernel(const float* __restrict__ kraw, const float* __restrict__ vraw,
                               const float* __restrict__ cs, const float* __restrict__ sn,
                               h16* __restrict__ Kh, h16* __restrict__ Vt) {
    long idx = (long)blockIdx.x * blockDim.x + threadIdx.x;  // B*S*KVH*HD
    if (idx >= (long)2 * 2048 * 4 * 128) return;
    int i  = (int)(idx & 127);
    int kh = (int)((idx >> 7) & 3);
    int s  = (int)((idx >> 9) & 2047);
    int b  = (int)(idx >> 20);
    long src = ((long)(b * 2048 + s)) * 512 + kh * 128 + i;
    float c = cs[s * 128 + i], sv = sn[s * 128 + i];
    float kv = kraw[src];
    float rot = (i < 64) ? -kraw[src + 64] : kraw[src - 64];
    Kh[(((long)b * 4 + kh) * 2048 + s) * 128 + i] = (h16)(kv * c + rot * sv);
    Vt[(((long)b * 4 + kh) * 128 + i) * 2048 + s] = (h16)vraw[src];
}

// ---------------------------------------------------------------------------
// Flash attention: one wave per (b, h, 16-row q tile). Causal, online softmax.
// ---------------------------------------------------------------------------
__global__ __launch_bounds__(32) void mla_attn(const h16* __restrict__ Qh,
                                               const h16* __restrict__ Kh,
                                               const h16* __restrict__ Vt,
                                               h16* __restrict__ Out) {
    __shared__ __attribute__((aligned(16))) h16 Ps[16 * LSTR];

    const int lane = threadIdx.x;
    const int r  = lane & 15;
    const int hi = lane >> 4;
    const int c0 = hi * 8;
    const int qt = blockIdx.x, h = blockIdx.y, b = blockIdx.z;
    const int kvh = h >> 2;
    const int qbase = qt * 16;
    const float scale = 0.08838834764831845f;  // 128^-0.5
    const float NEG_INF = -__builtin_inff();

    const h16* Qbase = Qh + (((long)b * 16 + h) * 2048) * 128;
    const h16* Kbase = Kh + (((long)b * 4 + kvh) * 2048) * 128;
    const h16* Vbase = Vt + (((long)b * 4 + kvh) * 128) * 2048;

    Frag qf[4];
#pragma unroll
    for (int ks = 0; ks < 4; ++ks) {
        const h16* rowp = Qbase + (long)(qbase + r) * 128 + ks * 32;
        ulonglong2 t0 = *(const ulonglong2*)(rowp + c0);
        ulonglong2 t1 = *(const ulonglong2*)(rowp + c0 + 16);
        qf[ks].q[0] = t0.x; qf[ks].q[1] = t0.y;
        qf[ks].q[2] = t1.x; qf[ks].q[3] = t1.y;
    }

    v8f o[8];
#pragma unroll
    for (int nt = 0; nt < 8; ++nt) o[nt] = v8f_zero();
    float mrow[8], lrow[8];
#pragma unroll
    for (int j = 0; j < 8; ++j) { mrow[j] = NEG_INF; lrow[j] = 0.0f; }

    const int kend = qbase + 16;
    for (int kb = 0; kb < kend; kb += 32) {
        v8f s0 = v8f_zero(), s1 = v8f_zero();
#pragma unroll
        for (int ks = 0; ks < 4; ++ks) {
            Frag kf0, kf1;
            const h16* k0p = Kbase + (long)(kb + r) * 128 + ks * 32;
            ulonglong2 a0 = *(const ulonglong2*)(k0p + c0);
            ulonglong2 a1 = *(const ulonglong2*)(k0p + c0 + 16);
            kf0.q[0] = a0.x; kf0.q[1] = a0.y; kf0.q[2] = a1.x; kf0.q[3] = a1.y;
            const h16* k1p = Kbase + (long)(kb + 16 + r) * 128 + ks * 32;
            ulonglong2 b0 = *(const ulonglong2*)(k1p + c0);
            ulonglong2 b1 = *(const ulonglong2*)(k1p + c0 + 16);
            kf1.q[0] = b0.x; kf1.q[1] = b0.y; kf1.q[2] = b1.x; kf1.q[3] = b1.y;
            s0 = __builtin_amdgcn_wmma_f32_16x16x32_f16(false, qf[ks].v, false, kf0.v,
                                                        (short)0, s0, false, false);
            s1 = __builtin_amdgcn_wmma_f32_16x16x32_f16(false, qf[ks].v, false, kf1.v,
                                                        (short)0, s1, false, false);
        }

        float alpha[8];
#pragma unroll
        for (int j = 0; j < 8; ++j) {
            int row = qbase + j + hi * 8;
            float a  = s0[j] * scale; if (kb + r      > row) a  = NEG_INF;
            float bb = s1[j] * scale; if (kb + 16 + r > row) bb = NEG_INF;
            float tm = fmaxf(a, bb);
            tm = fmaxf(tm, __shfl_xor(tm, 1, 16));
            tm = fmaxf(tm, __shfl_xor(tm, 2, 16));
            tm = fmaxf(tm, __shfl_xor(tm, 4, 16));
            tm = fmaxf(tm, __shfl_xor(tm, 8, 16));
            float mnew = fmaxf(mrow[j], tm);
            float al = __expf(mrow[j] - mnew);
            mrow[j] = mnew;
            float pa = __expf(a - mnew);
            float pb = __expf(bb - mnew);
            float rs = pa + pb;
            rs += __shfl_xor(rs, 1, 16);
            rs += __shfl_xor(rs, 2, 16);
            rs += __shfl_xor(rs, 4, 16);
            rs += __shfl_xor(rs, 8, 16);
            lrow[j] = lrow[j] * al + rs;
            alpha[j] = al;
            Ps[(j + hi * 8) * LSTR + r]      = (h16)pa;
            Ps[(j + hi * 8) * LSTR + 16 + r] = (h16)pb;
        }
#pragma unroll
        for (int nt = 0; nt < 8; ++nt)
#pragma unroll
            for (int j = 0; j < 8; ++j) o[nt][j] *= alpha[j];

        __syncthreads();

        Frag pf;
        {
            const h16* prow = Ps + r * LSTR;
            const unsigned long long* q0 = (const unsigned long long*)(prow + c0);
            const unsigned long long* q1 = (const unsigned long long*)(prow + c0 + 16);
            pf.q[0] = q0[0]; pf.q[1] = q0[1]; pf.q[2] = q1[0]; pf.q[3] = q1[1];
        }

#pragma unroll
        for (int nt = 0; nt < 8; ++nt) {
            Frag vf;
            const h16* vp = Vbase + (long)(nt * 16 + r) * 2048 + kb;
            ulonglong2 v0 = *(const ulonglong2*)(vp + c0);
            ulonglong2 v1 = *(const ulonglong2*)(vp + c0 + 16);
            vf.q[0] = v0.x; vf.q[1] = v0.y; vf.q[2] = v1.x; vf.q[3] = v1.y;
            o[nt] = __builtin_amdgcn_wmma_f32_16x16x32_f16(false, pf.v, false, vf.v,
                                                           (short)0, o[nt], false, false);
        }
        __syncthreads();
    }

#pragma unroll
    for (int j = 0; j < 8; ++j) {
        float inv = 1.0f / lrow[j];
        int srow = qbase + j + hi * 8;
        h16* orow = Out + ((long)b * 2048 + srow) * 2048 + h * 128;
#pragma unroll
        for (int nt = 0; nt < 8; ++nt)
            orow[nt * 16 + r] = (h16)(o[nt][j] * inv);
    }
}

// ---------------------------------------------------------------------------
// Host orchestration
// ---------------------------------------------------------------------------
extern "C" void kernel_launch(void* const* d_in, const int* in_sizes, int n_in,
                              void* d_out, int out_size, void* d_ws, size_t ws_size,
                              hipStream_t stream) {
    (void)in_sizes; (void)n_in; (void)out_size; (void)ws_size;
    const float* x    = (const float*)d_in[0];
    const float* cosb = (const float*)d_in[1];
    const float* sinb = (const float*)d_in[2];
    // d_in[3] = mask (unused; causal computed analytically)
    const float* W_q  = (const float*)d_in[4];
    const float* W_kv = (const float*)d_in[5];
    const float* W_k  = (const float*)d_in[6];
    const float* W_v  = (const float*)d_in[7];
    const float* W_o  = (const float*)d_in[8];

    const long BS = 4096;  // B*S
    char* ws = (char*)d_ws;
    size_t off = 0;
    auto alloc = [&](size_t bytes) -> void* {
        void* p = ws + off;
        off += (bytes + 255) & ~(size_t)255;
        return p;
    };
    h16*   xb     = (h16*)alloc(BS * 2048 * 2);
    h16*   Wqt    = (h16*)alloc(2048L * 2048 * 2);
    h16*   Wkvt   = (h16*)alloc(512L * 2048 * 2);
    h16*   Wkt    = (h16*)alloc(512L * 512 * 2);
    h16*   Wvt    = (h16*)alloc(512L * 512 * 2);
    h16*   Wot    = (h16*)alloc(2048L * 2048 * 2);
    float* q_raw  = (float*)alloc(BS * 2048 * 4);
    h16*   latent = (h16*)alloc(BS * 512 * 2);
    float* k_raw  = (float*)alloc(BS * 512 * 4);
    float* v_raw  = (float*)alloc(BS * 512 * 4);
    h16*   Qh     = (h16*)alloc(2L * 16 * 2048 * 128 * 2);
    h16*   Kh     = (h16*)alloc(2L * 4 * 2048 * 128 * 2);
    h16*   Vt     = (h16*)alloc(2L * 4 * 128 * 2048 * 2);
    h16*   attn   = (h16*)alloc(BS * 2048 * 2);

    cast_f32_h16<<<(BS * 2048 + 255) / 256, 256, 0, stream>>>(x, xb, BS * 2048);
    transpose_f32_h16<<<(2048L * 2048 + 255) / 256, 256, 0, stream>>>(W_q,  Wqt,  2048, 2048);
    transpose_f32_h16<<<(2048L * 512  + 255) / 256, 256, 0, stream>>>(W_kv, Wkvt, 2048, 512);
    transpose_f32_h16<<<(512L  * 512  + 255) / 256, 256, 0, stream>>>(W_k,  Wkt,  512,  512);
    transpose_f32_h16<<<(512L  * 512  + 255) / 256, 256, 0, stream>>>(W_v,  Wvt,  512,  512);
    transpose_f32_h16<<<(2048L * 2048 + 255) / 256, 256, 0, stream>>>(W_o,  Wot,  2048, 2048);

    gemm_wmma_f16<true><<<dim3(4096 / 128, 2048 / 128), 256, 0, stream>>>(
        xb, Wqt, q_raw, 4096, 2048, 2048);
    gemm_wmma_f16<false><<<dim3(4096 / 128, 512 / 128), 256, 0, stream>>>(
        xb, Wkvt, latent, 4096, 512, 2048);
    gemm_wmma_f16<true><<<dim3(4096 / 128, 512 / 128), 256, 0, stream>>>(
        latent, Wkt, k_raw, 4096, 512, 512);
    gemm_wmma_f16<true><<<dim3(4096 / 128, 512 / 128), 256, 0, stream>>>(
        latent, Wvt, v_raw, 4096, 512, 512);

    rope_q_kernel<<<(2L * 2048 * 16 * 128 + 255) / 256, 256, 0, stream>>>(
        q_raw, cosb, sinb, Qh);
    rope_kv_kernel<<<(2L * 2048 * 4 * 128 + 255) / 256, 256, 0, stream>>>(
        k_raw, v_raw, cosb, sinb, Kh, Vt);

    mla_attn<<<dim3(2048 / 16, 16, 2), 32, 0, stream>>>(Qh, Kh, Vt, attn);

    gemm_wmma_f16<true><<<dim3(4096 / 128, 2048 / 128), 256, 0, stream>>>(
        attn, Wot, (float*)d_out, 4096, 2048, 2048);
}
